// Seq2Seq_4363686773050
// MI455X (gfx1250) — matrix operand order repaired and checked by reading, not compiled, not run on hardware
//
#include <hip/hip_runtime.h>
#include <hip/hip_bf16.h>
#include <math.h>

typedef _Float16 h16;
typedef __attribute__((ext_vector_type(16))) _Float16 v16h;
typedef __attribute__((ext_vector_type(8)))  _Float16 v8h;
typedef __attribute__((ext_vector_type(8)))  float    v8f;

#define DEVI static __device__ __forceinline__

namespace cfg {
constexpr int H = 64, L = 64, D = 128, B = 128, T = 256;
constexpr int G = 192;                 // 3*H gate columns
constexpr int RING = 8;                // ring depth (timesteps) per layer boundary
constexpr size_t RING_LAYER_ELEMS = (size_t)RING * B * H;          // f16 elements
constexpr size_t RING_OFF   = 8192;                                 // bytes into ws
constexpr size_t DECOUT_OFF = RING_OFF + (size_t)L * RING_LAYER_ELEMS * sizeof(h16);
constexpr int SMEM_BYTES = 183296;     // worst case (layer 0): weights f16 + biases + h f32
}

// ---------------- WMMA helpers (CDNA5 gfx1250, wave32) ----------------

DEVI v8f wmma_f16(v16h a, v16h b, v8f c) {
  // D = A(16x32 f16) * B(32x16 f16) + C(16x16 f32)
  return __builtin_amdgcn_wmma_f32_16x16x32_f16(false, a, false, b, (short)0, c,
                                                false, false);
}

// A fragment (16x32, M x K) from row-major f16 memory.
// ISA layout: lane L -> row M=L%16; lanes 0-15 hold K=[0..7],[16..23],
// lanes 16-31 hold K=[8..15],[24..31] (2 f16 per VGPR, ascending K).
DEVI v16h loadA16(const h16* base, int ld, int row16, int k0, int lane) {
  const int r  = row16 + (lane & 15);
  const int ka = k0 + ((lane >> 4) << 3);
  union { v16h v; v8h h[2]; } u;
  u.h[0] = *(const v8h*)(base + (size_t)r * ld + ka);
  u.h[1] = *(const v8h*)(base + (size_t)r * ld + ka + 16);
  return u.v;
}

// Same A fragment from f32 memory with on-the-fly f32->f16 conversion.
DEVI v16h loadA32(const float* base, size_t ld, int row16, int k0, int lane) {
  const int r  = row16 + (lane & 15);
  const int ka = k0 + ((lane >> 4) << 3);
  const float* p0 = base + (size_t)r * ld + ka;
  union { v16h v; h16 e[16]; } u;
#pragma unroll
  for (int i = 0; i < 8; ++i) u.e[i] = (h16)p0[i];
#pragma unroll
  for (int i = 0; i < 8; ++i) u.e[8 + i] = (h16)p0[16 + i];
  return u.v;
}

// B fragment (32x16, K x N). Weight W stored row-major as (N, K) so that
// B[k][n] = W[n][k]: lane L -> col N=L%16; lanes 0-15 hold K=[0..7],[16..23],
// lanes 16-31 hold K=[8..15],[24..31] (matches A) -> two aligned 16-byte reads.
DEVI v16h loadB16(const h16* w, int ldk, int n16, int k0, int lane) {
  const int n  = n16 + (lane & 15);
  const int ks = k0 + ((lane >> 4) << 3);
  union { v16h v; v8h h[2]; } u;
  u.h[0] = *(const v8h*)(w + (size_t)n * ldk + ks);
  u.h[1] = *(const v8h*)(w + (size_t)n * ldk + ks + 16);
  return u.v;
}

DEVI float sigmoidf_(float x) { return 1.f / (1.f + __expf(-x)); }

// ---------------- persistent layer-pipelined GRU kernel ----------------
// grid = 64 blocks (one per layer, one per WGP), 256 threads = 8 wave32.
// Wave m owns batch rows [16m, 16m+16): its A-fragments, gate tiles and
// state rows are wave-private -> no intra-block barriers in steady state.

__global__ void __launch_bounds__(256, 1)
gru_pipeline_kernel(const float* __restrict__ ctx,
                    const float* __restrict__ eWih0, const float* __restrict__ eWih,
                    const float* __restrict__ eWhh,  const float* __restrict__ eBih,
                    const float* __restrict__ eBhh,
                    const float* __restrict__ dWih0, const float* __restrict__ dWih,
                    const float* __restrict__ dWhh,  const float* __restrict__ dBih,
                    const float* __restrict__ dBhh,
                    h16* __restrict__ ring, h16* __restrict__ decout,
                    unsigned* __restrict__ ready, unsigned* __restrict__ done) {
  using namespace cfg;
  __shared__ __align__(16) unsigned char smem[SMEM_BYTES];
  const int l    = blockIdx.x;
  const int tid  = threadIdx.x;
  const int m    = tid >> 5;          // wave id 0..7
  const int lane = tid & 31;
  const int Kin  = (l == 0) ? D : H;  // ih-GEMM reduction depth

  // ---- LDS carve-out: enc/dec weights (f16), biases (f32), hidden state (f32)
  h16* sWihE = (h16*)smem;            // G x Kin
  h16* sWhhE = sWihE + G * Kin;       // G x H
  h16* sWihD = sWhhE + G * H;         // G x Kin
  h16* sWhhD = sWihD + G * Kin;       // G x H
  float* sBihE = (float*)(sWhhD + G * H);
  float* sBhhE = sBihE + G;
  float* sBihD = sBhhE + G;
  float* sBhhD = sBihD + G;
  float* sHst  = sBhhD + G;           // B x H hidden state, fp32

  // ---- one-time: convert weights fp32->f16 into LDS, zero hidden state
  const float* gWihE = (l == 0) ? eWih0 : eWih + (size_t)(l - 1) * G * H;
  const float* gWihD = (l == 0) ? dWih0 : dWih + (size_t)(l - 1) * G * H;
  const float* gWhhE = eWhh + (size_t)l * G * H;
  const float* gWhhD = dWhh + (size_t)l * G * H;
  for (int i = tid; i < G * Kin; i += 256) { sWihE[i] = (h16)gWihE[i]; sWihD[i] = (h16)gWihD[i]; }
  for (int i = tid; i < G * H;   i += 256) { sWhhE[i] = (h16)gWhhE[i]; sWhhD[i] = (h16)gWhhD[i]; }
  for (int i = tid; i < G; i += 256) {
    sBihE[i] = eBih[(size_t)l * G + i]; sBhhE[i] = eBhh[(size_t)l * G + i];
    sBihD[i] = dBih[(size_t)l * G + i]; sBhhD[i] = dBhh[(size_t)l * G + i];
  }
  for (int i = tid; i < B * H; i += 256) sHst[i] = 0.f;
  __syncthreads();

  unsigned* rdyPrev = (l > 0)     ? &ready[(l - 1) * 8 + m] : nullptr;
  unsigned* rdyMine = &ready[l * 8 + m];
  unsigned* dnMine  = &done[l * 8 + m];
  unsigned* dnNext  = (l < L - 1) ? &done[(l + 1) * 8 + m] : nullptr;
  const int rowb = m * 16;

  const h16* ringIn  = (l > 0)     ? ring + (size_t)(l - 1) * RING_LAYER_ELEMS : nullptr;
  h16*       ringOut = (l < L - 1) ? ring + (size_t)l * RING_LAYER_ELEMS : nullptr;

  // 512-step pipeline: t<T = encoder, t>=T = decoder. Decoder h0 == encoder
  // final h, which is exactly what sHst holds at t==T. Flags are monotonic
  // published/consumed step counters per (layer, wave).
  for (int t = 0; t < 2 * T; ++t) {
    const bool dec = (t >= T);
    const int  tt  = dec ? t - T : t;
    const h16*   Wih = dec ? sWihD : sWihE;
    const h16*   Whh = dec ? sWhhD : sWhhE;
    const float* Bih = dec ? sBihD : sBihE;
    const float* Bhh = dec ? sBhhD : sBhhE;

    v8f acc[12], hacc[4];               // acc: r/z gates get ih+hh fused; n-gate ih
    const v8f z8 = {0.f, 0.f, 0.f, 0.f, 0.f, 0.f, 0.f, 0.f};
#pragma unroll
    for (int n = 0; n < 12; ++n) acc[n] = z8;
#pragma unroll
    for (int n = 0; n < 4; ++n) hacc[n] = z8;

    // ---- gi = x @ Wih^T ----
    if (l > 0) {
      while (__hip_atomic_load(rdyPrev, __ATOMIC_ACQUIRE, __HIP_MEMORY_SCOPE_AGENT)
             <= (unsigned)t)
        __builtin_amdgcn_s_sleep(1);
      const h16* x = ringIn + (size_t)(t % RING) * B * H;
      for (int kk = 0; kk < H; kk += 32) {
        v16h a = loadA16(x, H, rowb, kk, lane);
#pragma unroll
        for (int n = 0; n < 12; ++n)
          acc[n] = wmma_f16(a, loadB16(Wih, H, n * 16, kk, lane), acc[n]);
      }
      if (lane == 0)   // input fully consumed -> release ring slot
        __hip_atomic_store(dnMine, (unsigned)(t + 1), __ATOMIC_RELEASE,
                           __HIP_MEMORY_SCOPE_AGENT);
    } else {
      // layer 0: encoder reads context[:,tt,:]; decoder teacher-forces
      // context[:,tt-1,:] (zeros at tt==0, leaving acc at 0).
      const float* x = nullptr;
      if (!dec)        x = ctx + (size_t)tt * D;
      else if (tt > 0) x = ctx + (size_t)(tt - 1) * D;
      if (x) {
        for (int kk = 0; kk < D; kk += 32) {
          v16h a = loadA32(x, (size_t)T * D, rowb, kk, lane);
#pragma unroll
          for (int n = 0; n < 12; ++n)
            acc[n] = wmma_f16(a, loadB16(Wih, D, n * 16, kk, lane), acc[n]);
        }
      }
    }

    // ---- gh = h @ Whh^T  (r/z thirds fused into acc; n-third kept apart) ----
    for (int kk = 0; kk < H; kk += 32) {
      v16h a = loadA32(sHst, H, rowb, kk, lane);
#pragma unroll
      for (int n = 0; n < 8; ++n)
        acc[n] = wmma_f16(a, loadB16(Whh, H, n * 16, kk, lane), acc[n]);
#pragma unroll
      for (int n = 0; n < 4; ++n)
        hacc[n] = wmma_f16(a, loadB16(Whh, H, (8 + n) * 16, kk, lane), hacc[n]);
    }

    // ---- back-pressure: don't overwrite a ring slot the consumer hasn't read
    if (l < L - 1 && t >= RING) {
      const unsigned need = (unsigned)(t + 1 - RING);
      while (__hip_atomic_load(dnNext, __ATOMIC_ACQUIRE, __HIP_MEMORY_SCOPE_AGENT) < need)
        __builtin_amdgcn_s_sleep(1);
    }
    h16* xout = (l < L - 1) ? ringOut + (size_t)(t % RING) * B * H : nullptr;

    // ---- gates + state update (C layout: lane L, vgpr r -> M=r+8*(L/16), N=L%16)
    const int nl   = lane & 15;
    const int rsel = lane >> 4;
#pragma unroll
    for (int c = 0; c < 4; ++c) {
      const int col  = c * 16 + nl;
      const float b_r  = Bih[col]      + Bhh[col];
      const float b_z  = Bih[64 + col] + Bhh[64 + col];
      const float bi_n = Bih[128 + col];
      const float bh_n = Bhh[128 + col];
#pragma unroll
      for (int r = 0; r < 8; ++r) {
        const int row = rowb + r + 8 * rsel;
        const float rg = sigmoidf_(acc[c][r]     + b_r);
        const float zg = sigmoidf_(acc[c + 4][r] + b_z);
        const float hn = hacc[c][r] + bh_n;
        const float ng = tanhf(acc[c + 8][r] + bi_n + rg * hn);
        const float hp = sHst[(size_t)row * H + col];
        const float hv = (1.f - zg) * ng + zg * hp;
        sHst[(size_t)row * H + col] = hv;
        if (l < L - 1)  xout[(size_t)row * H + col] = (h16)hv;
        else if (dec)   decout[((size_t)row * T + tt) * H + col] = (h16)hv;
      }
    }

    if (l < L - 1 && lane == 0)
      __hip_atomic_store(rdyMine, (unsigned)(t + 1), __ATOMIC_RELEASE,
                         __HIP_MEMORY_SCOPE_AGENT);
  }
}

// ---------------- output projection: (B*T x H) @ Wo^T + bo ----------------
// grid = 256 blocks x 256 threads; block handles 128 rows, wave m -> 16 rows.

__global__ void __launch_bounds__(256, 2)
proj_kernel(const h16* __restrict__ decout, const float* __restrict__ Wo,
            const float* __restrict__ bo, float* __restrict__ out) {
  using namespace cfg;
  __shared__ __align__(16) h16 sW[D * H];  // Wo (D x H) as f16, row-major (N,K)
  __shared__ float sB[D];
  const int tid  = threadIdx.x;
  const int m    = tid >> 5;
  const int lane = tid & 31;
  for (int i = tid; i < D * H; i += 256) sW[i] = (h16)Wo[i];
  for (int i = tid; i < D; i += 256) sB[i] = bo[i];
  __syncthreads();

  const int rowb = blockIdx.x * 128 + m * 16;
  v8f acc[8];
  const v8f z8 = {0.f, 0.f, 0.f, 0.f, 0.f, 0.f, 0.f, 0.f};
#pragma unroll
  for (int n = 0; n < 8; ++n) acc[n] = z8;
  for (int kk = 0; kk < H; kk += 32) {
    v16h a = loadA16(decout, H, rowb, kk, lane);
#pragma unroll
    for (int n = 0; n < 8; ++n)
      acc[n] = wmma_f16(a, loadB16(sW, H, n * 16, kk, lane), acc[n]);
  }
  const int nl = lane & 15, rsel = lane >> 4;
#pragma unroll
  for (int n = 0; n < 8; ++n) {
    const int col = n * 16 + nl;
    const float b = sB[col];
#pragma unroll
    for (int r = 0; r < 8; ++r) {
      const int row = rowb + r + 8 * rsel;
      out[(size_t)row * D + col] = acc[n][r] + b;
    }
  }
}

__global__ void init_flags_kernel(unsigned* f) { f[threadIdx.x] = 0u; }

// ---------------- launch ----------------

extern "C" void kernel_launch(void* const* d_in, const int* in_sizes, int n_in,
                              void* d_out, int out_size, void* d_ws, size_t ws_size,
                              hipStream_t stream) {
  using namespace cfg;
  (void)in_sizes; (void)n_in; (void)out_size; (void)ws_size;
  const float* ctx   = (const float*)d_in[0];
  const float* eWih0 = (const float*)d_in[1];
  const float* eWih  = (const float*)d_in[2];
  const float* eWhh  = (const float*)d_in[3];
  const float* eBih  = (const float*)d_in[4];
  const float* eBhh  = (const float*)d_in[5];
  const float* dWih0 = (const float*)d_in[6];
  const float* dWih  = (const float*)d_in[7];
  const float* dWhh  = (const float*)d_in[8];
  const float* dBih  = (const float*)d_in[9];
  const float* dBhh  = (const float*)d_in[10];
  const float* Wo    = (const float*)d_in[11];
  const float* bo    = (const float*)d_in[12];

  unsigned char* ws = (unsigned char*)d_ws;
  unsigned* flags = (unsigned*)ws;                   // ready[512] | done[512]
  h16* ring   = (h16*)(ws + RING_OFF);               // 64 x RING x B x H f16 (8 MB)
  h16* decout = (h16*)(ws + DECOUT_OFF);             // (B*T) x H f16 (4 MB)
  float* out  = (float*)d_out;

  init_flags_kernel<<<dim3(1), dim3(1024), 0, stream>>>(flags);
  gru_pipeline_kernel<<<dim3(L), dim3(256), 0, stream>>>(
      ctx, eWih0, eWih, eWhh, eBih, eBhh,
      dWih0, dWih, dWhh, dBih, dBhh,
      ring, decout, flags, flags + 512);
  proj_kernel<<<dim3((B * T) / 128), dim3(256), 0, stream>>>(decout, Wo, bo, out);
}
